// EquivariantMessagePasser_28922309771659
// MI455X (gfx1250) — matrix operand order repaired and checked by reading, not compiled, not run on hardware
//
#include <hip/hip_runtime.h>
#include <hip/hip_bf16.h>

typedef __attribute__((ext_vector_type(16))) __bf16 v16bf;
typedef __attribute__((ext_vector_type(8)))  float  v8f;

#define NATOMS 10000
#define NEDGES 100000
#define CUTOFF 5.0f
// per-l constants: KMAX={128,96,64,32}, lo-slice={96,64,32,0}, 3^l={1,3,9,27}
// uncoupled layout per atom: 40 rows x 32 ch  (row offsets per l: 0,1,4,13)
#define UNC_PER_ATOM 1280
#define FEAT_PER_ATOM 960

// ---------------------------------------------------------------- zero
__global__ void k_zero(float* __restrict__ p, int n) {
    int i = blockIdx.x * blockDim.x + threadIdx.x;
    if (i < n) p[i] = 0.0f;
}

// ------------------------------------------------- per-atom uncouple of feat
// uncF[n, row, k]  (row in 0..39, k in 0..31), f32
__global__ void k_atom_uncouple(const float* __restrict__ f0, const float* __restrict__ f1,
                                const float* __restrict__ f2, const float* __restrict__ f3,
                                const float* __restrict__ U0, const float* __restrict__ U1,
                                const float* __restrict__ U2, const float* __restrict__ U3,
                                float* __restrict__ uncF) {
    __shared__ float fs[FEAT_PER_ATOM];
    const int n = blockIdx.x, tid = threadIdx.x;
    for (int i = tid; i < 128; i += 256) fs[i]       = f0[(size_t)n * 128 + i];
    for (int i = tid; i < 288; i += 256) fs[128 + i] = f1[(size_t)n * 288 + i];
    for (int i = tid; i < 320; i += 256) fs[416 + i] = f2[(size_t)n * 320 + i];
    for (int i = tid; i < 224; i += 256) fs[736 + i] = f3[(size_t)n * 224 + i];
    __syncthreads();
    const float* Us[4] = {U0, U1, U2, U3};
    const int FOFF[4] = {0, 128, 416, 736};
    const int KM[4]   = {128, 96, 64, 32};
    const int LO[4]   = {96, 64, 32, 0};
    for (int idx = tid; idx < UNC_PER_ATOM; idx += 256) {
        int row = idx >> 5, k = idx & 31;
        int l, d;
        if (row < 1)       { l = 0; d = row; }
        else if (row < 4)  { l = 1; d = row - 1; }
        else if (row < 13) { l = 2; d = row - 4; }
        else               { l = 3; d = row - 13; }
        const int M = (l + 1) * (l + 1);
        const float* U = Us[l];
        float acc = 0.0f;
        for (int m = 0; m < M; ++m) {
            int lp = (m < 1) ? 0 : (m < 4) ? 1 : (m < 9) ? 2 : 3;
            int mm = m - lp * lp;
            acc += U[d * M + m] * fs[FOFF[lp] + mm * KM[lp] + LO[l] + k];
        }
        uncF[(size_t)n * UNC_PER_ATOM + idx] = acc;
    }
}

// ----------------------------------------------------- edge stage, wave/edge
__global__ void __launch_bounds__(256)
k_edge(const float* __restrict__ r,
       const float* __restrict__ sh0, const float* __restrict__ sh1,
       const float* __restrict__ sh2, const float* __restrict__ sh3,
       const float* __restrict__ Wr0, const float* __restrict__ Wr1,
       const float* __restrict__ Wr2, const float* __restrict__ Wr3,
       const float* __restrict__ U0, const float* __restrict__ U1,
       const float* __restrict__ U2, const float* __restrict__ U3,
       const int* __restrict__ centers, const int* __restrict__ neighbors,
       const float* __restrict__ uncF, float* __restrict__ pooled) {
    const int gid  = blockIdx.x * blockDim.x + threadIdx.x;
    const int e    = gid >> 5;
    const int lane = gid & 31;
    if (e >= NEDGES) return;

    const int nbr = neighbors[e], ctr = centers[e];
    const float* uf = uncF + (size_t)nbr * UNC_PER_ATOM;
    float* pl = pooled + (size_t)ctr * UNC_PER_ATOM;
    // warm the whole 5120B gather region (L2-resident table) while we do math
    __builtin_prefetch(uf + (size_t)lane * 40, 0, 0);

    const float rv  = r[e];
    const float x   = rv * (3.14159265358979f / CUTOFF);
    float t = rv * (1.0f / CUTOFF);
    t = fminf(fmaxf(t, 0.0f), 1.0f);
    const float fc  = 0.5f * (__cosf(3.14159265358979f * t) + 1.0f);
    const float inv = fc / (rv + 1e-6f);
    float rb[8];
#pragma unroll
    for (int j = 0; j < 8; ++j) rb[j] = __sinf((float)(j + 1) * x) * inv;

    float s[16];
    s[0] = sh0[e];
#pragma unroll
    for (int i = 0; i < 3; ++i) s[1 + i] = sh1[(size_t)e * 3 + i];
#pragma unroll
    for (int i = 0; i < 5; ++i) s[4 + i] = sh2[(size_t)e * 5 + i];
#pragma unroll
    for (int i = 0; i < 7; ++i) s[9 + i] = sh3[(size_t)e * 7 + i];

    const float* Us[4] = {U0, U1, U2, U3};
    const float* Wr[4] = {Wr0, Wr1, Wr2, Wr3};
    constexpr int KM[4] = {128, 96, 64, 32};
    constexpr int LO[4] = {96, 64, 32, 0};
    constexpr int P3[4] = {1, 3, 9, 27};
    constexpr int RO[4] = {0, 1, 4, 13};

#pragma unroll
    for (int l = 0; l < 4; ++l) {
        const int lo = LO[l];
        float g[16];
#pragma unroll
        for (int lp = 0; lp <= l; ++lp) {
            float rad = 0.0f;
            const float* W = Wr[lp];
#pragma unroll
            for (int j = 0; j < 8; ++j) rad += rb[j] * W[j * KM[lp] + lo + lane];
#pragma unroll
            for (int mm = 0; mm < 2 * lp + 1; ++mm) g[lp * lp + mm] = s[lp * lp + mm] * rad;
        }
        const int M = (l + 1) * (l + 1);
        const float* U = Us[l];
        for (int d = 0; d < P3[l]; ++d) {
            float acc = 0.0f;
#pragma unroll
            for (int m = 0; m < M; ++m) acc += U[d * M + m] * g[m];
            const int off = (RO[l] + d) * 32 + lane;
            atomicAdd(&pl[off], acc * uf[off]);
        }
    }
}

// ------------------------------------- per-atom couple + concat, emit bf16 cat
__global__ void k_couple(const float* __restrict__ U0, const float* __restrict__ U1,
                         const float* __restrict__ U2, const float* __restrict__ U3,
                         const float* __restrict__ pooled, __bf16* __restrict__ cat) {
    __shared__ float ps[UNC_PER_ATOM];
    const int n = blockIdx.x, tid = threadIdx.x;
    for (int i = tid; i < UNC_PER_ATOM; i += 256) ps[i] = pooled[(size_t)n * UNC_PER_ATOM + i];
    __syncthreads();
    const float* Us[4] = {U0, U1, U2, U3};
    const int BASE[4] = {0, 128, 416, 736};
    const int KM[4]   = {128, 96, 64, 32};
    const int RO[4]   = {0, 1, 4, 13};
    const int P3[4]   = {1, 3, 9, 27};
    const size_t CATO[4] = {0, 1280000, 4160000, 7360000};  // per-l [R,K] matrices
    for (int idx = tid; idx < FEAT_PER_ATOM; idx += 256) {
        const int l   = (idx < 128) ? 0 : (idx < 416) ? 1 : (idx < 736) ? 2 : 3;
        const int rel = idx - BASE[l];
        const int mm  = rel / KM[l];
        const int c   = rel % KM[l];
        const int lp  = l + (c >> 5);
        const int k   = c & 31;
        const int m   = l * l + mm;
        const int Mlp = (lp + 1) * (lp + 1);
        const float* U = Us[lp];
        float acc = 0.0f;
        for (int d = 0; d < P3[lp]; ++d) acc += U[d * Mlp + m] * ps[(RO[lp] + d) * 32 + k];
        cat[CATO[l] + ((size_t)n * (2 * l + 1) + mm) * KM[l] + c] = (__bf16)acc;
    }
}

// ------------------- Wlin f32 -> bf16, pre-swizzled into WMMA B-fragment order
// Bfrag layout: [ks][tn][lane][idx0..15] bf16, lane = (kk>>4)*16 + (n&15),
// idx = kk&15, where k = ks*32+kk, tn = n>>4.  One wave-lane's 16 B values are
// then contiguous (two 16B vector loads).
__global__ void k_wfrag(const float* __restrict__ W, __bf16* __restrict__ Bf, int Kdim) {
    int i = blockIdx.x * blockDim.x + threadIdx.x;
    if (i >= Kdim * Kdim) return;
    const int k = i / Kdim, n = i % Kdim;
    const int Ntiles = Kdim >> 4;
    const int ks = k >> 5, kk = k & 31;
    const int tn = n >> 4, lr = n & 15;
    const int lane = ((kk >> 4) << 4) | lr;
    const int idx  = kk & 15;
    Bf[((size_t)(ks * Ntiles + tn) * 32 + lane) * 16 + idx] = (__bf16)W[i];
}

// ------------------------------------- WMMA GEMM: Out = Fin + A(bf16) @ B(bf16)
// A: [Rtiles*16, ld] row-major bf16; Bf: fragment-ordered bf16 (whole matrix,
// KSTEPS*NT*512 elements) staged in LDS per block.  One wave computes a
// 16 x (16*NTB) strip: all A fragments preloaded to VGPRs (one clause, one
// wait), B read via ds_load_b128 at immediate offsets, NTB accumulators.
template <int NTB, int KSTEPS, int NT>
__global__ void __launch_bounds__(256)
k_gemm_wmma(const __bf16* __restrict__ A, const __bf16* __restrict__ Bf,
            const float* __restrict__ Fin, float* __restrict__ Out,
            int Rtiles, int NGroups, int ld) {
    __shared__ __bf16 Bs[KSTEPS * NT * 512];
    const int tid = threadIdx.x;
    // cooperative stage of the whole fragment-ordered B matrix (<=32KB)
    {
        const uint4* src = reinterpret_cast<const uint4*>(Bf);
        uint4* dst = reinterpret_cast<uint4*>(Bs);
        constexpr int NV = KSTEPS * NT * 512 / 8;   // uint4 = 8 bf16
#pragma unroll
        for (int i = tid; i < NV; i += 256) dst[i] = src[i];
    }
    __syncthreads();

    const int wave = (blockIdx.x * blockDim.x + tid) >> 5;
    const bool active = wave < Rtiles * NGroups;
    const int tm  = wave / NGroups;
    const int tn0 = (wave % NGroups) * NTB;
    const int lane = tid & 31;
    const int half = lane >> 4;      // K-half select
    const int lr   = lane & 15;      // A: row, D: col%16
    if (!active) return;

    // preload all A fragments: lane's values are K = {k0+16h..+15}
    const __bf16* Arow = A + (size_t)(tm * 16 + lr) * ld + 16 * half;
    v16bf a[KSTEPS];
#pragma unroll
    for (int ks = 0; ks < KSTEPS; ++ks) {
        *(reinterpret_cast<uint4*>(&a[ks]) + 0) =
            *reinterpret_cast<const uint4*>(Arow + ks * 32);
        *(reinterpret_cast<uint4*>(&a[ks]) + 1) =
            *reinterpret_cast<const uint4*>(Arow + ks * 32 + 8);
    }

    v8f c[NTB];
#pragma unroll
    for (int t = 0; t < NTB; ++t) c[t] = {0.f, 0.f, 0.f, 0.f, 0.f, 0.f, 0.f, 0.f};

    const __bf16* Bbase = Bs + ((size_t)tn0 * 32 + lane) * 16;
#pragma unroll
    for (int ks = 0; ks < KSTEPS; ++ks) {
#pragma unroll
        for (int t = 0; t < NTB; ++t) {
            const __bf16* bp = Bbase + (ks * NT + t) * 512;   // immediate DS offsets
            v16bf b;
            *(reinterpret_cast<uint4*>(&b) + 0) = *reinterpret_cast<const uint4*>(bp);
            *(reinterpret_cast<uint4*>(&b) + 1) = *reinterpret_cast<const uint4*>(bp + 8);
            c[t] = __builtin_amdgcn_wmma_f32_16x16x32_bf16(false, a[ks], false, b, (short)0,
                                                           c[t], false, false);
        }
    }
    // C/D: VGPR v -> row v (lanes0-15) / v+8 (lanes16-31); lane%16 -> col
#pragma unroll
    for (int t = 0; t < NTB; ++t) {
        const int col = (tn0 + t) * 16 + lr;
#pragma unroll
        for (int v = 0; v < 8; ++v) {
            const int row = tm * 16 + v + 8 * half;
            const size_t o = (size_t)row * ld + col;
            Out[o] = Fin[o] + c[t][v];
        }
    }
}

extern "C" void kernel_launch(void* const* d_in, const int* in_sizes, int n_in,
                              void* d_out, int out_size, void* d_ws, size_t ws_size,
                              hipStream_t stream) {
    // dict order: r, then per l: sh,feat,Wrad,U,Wlin, then centers, neighbors
    const float* r = (const float*)d_in[0];
    const float* sh[4]; const float* ft[4]; const float* Wr[4];
    const float* U[4];  const float* Wl[4];
    for (int l = 0; l < 4; ++l) {
        sh[l] = (const float*)d_in[1 + 5 * l + 0];
        ft[l] = (const float*)d_in[1 + 5 * l + 1];
        Wr[l] = (const float*)d_in[1 + 5 * l + 2];
        U[l]  = (const float*)d_in[1 + 5 * l + 3];
        Wl[l] = (const float*)d_in[1 + 5 * l + 4];
    }
    const int* centers   = (const int*)d_in[21];
    const int* neighbors = (const int*)d_in[22];
    float* out = (float*)d_out;

    char* ws = (char*)d_ws;
    float*  uncF   = (float*)ws;                                           // 51.2 MB
    float*  pooled = (float*)(ws + (size_t)NATOMS * UNC_PER_ATOM * 4);     // 51.2 MB
    __bf16* catbf  = (__bf16*)(ws + (size_t)NATOMS * UNC_PER_ATOM * 8);    // 19.2 MB
    __bf16* wlbf   = (__bf16*)(ws + (size_t)NATOMS * UNC_PER_ATOM * 8
                                  + (size_t)NATOMS * FEAT_PER_ATOM * 2);   // 60 KB

    {   // pooled must be zero every call (deterministic accumulation)
        const int n = NATOMS * UNC_PER_ATOM;
        k_zero<<<(n + 255) / 256, 256, 0, stream>>>(pooled, n);
    }
    k_atom_uncouple<<<NATOMS, 256, 0, stream>>>(ft[0], ft[1], ft[2], ft[3],
                                                U[0], U[1], U[2], U[3], uncF);
    k_edge<<<(NEDGES * 32 + 255) / 256, 256, 0, stream>>>(
        r, sh[0], sh[1], sh[2], sh[3], Wr[0], Wr[1], Wr[2], Wr[3],
        U[0], U[1], U[2], U[3], centers, neighbors, uncF, pooled);
    k_couple<<<NATOMS, 256, 0, stream>>>(U[0], U[1], U[2], U[3], pooled, catbf);

    const size_t CATO[4] = {0, 1280000, 4160000, 7360000};
    const size_t WO[4]   = {0, 16384, 25600, 29696};
    const int    KM[4]   = {128, 96, 64, 32};
    for (int l = 0; l < 4; ++l) {
        const int n = KM[l] * KM[l];
        k_wfrag<<<(n + 255) / 256, 256, 0, stream>>>(Wl[l], wlbf + WO[l], KM[l]);
    }
    // per-l tiling: NTB (n-tiles/wave), KSTEPS (=K/32), NT (=K/16)
    {   // l=0: K=128, Rt=625, Nt=8 -> NTB=4, 2 groups
        const int Rt = 625, NG = 2;
        const int blocks = (Rt * NG + 7) / 8;
        k_gemm_wmma<4, 4, 8><<<blocks, 256, 0, stream>>>(catbf + CATO[0], wlbf + WO[0],
                                                         ft[0], out + CATO[0], Rt, NG, 128);
    }
    {   // l=1: K=96, Rt=1875, Nt=6 -> NTB=3, 2 groups
        const int Rt = 1875, NG = 2;
        const int blocks = (Rt * NG + 7) / 8;
        k_gemm_wmma<3, 3, 6><<<blocks, 256, 0, stream>>>(catbf + CATO[1], wlbf + WO[1],
                                                         ft[1], out + CATO[1], Rt, NG, 96);
    }
    {   // l=2: K=64, Rt=3125, Nt=4 -> NTB=4, 1 group
        const int Rt = 3125, NG = 1;
        const int blocks = (Rt * NG + 7) / 8;
        k_gemm_wmma<4, 2, 4><<<blocks, 256, 0, stream>>>(catbf + CATO[2], wlbf + WO[2],
                                                         ft[2], out + CATO[2], Rt, NG, 64);
    }
    {   // l=3: K=32, Rt=4375, Nt=2 -> NTB=2, 1 group
        const int Rt = 4375, NG = 1;
        const int blocks = (Rt * NG + 7) / 8;
        k_gemm_wmma<2, 1, 2><<<blocks, 256, 0, stream>>>(catbf + CATO[3], wlbf + WO[3],
                                                         ft[3], out + CATO[3], Rt, NG, 32);
    }
}
